// SwapGNN_16484084483262
// MI455X (gfx1250) — matrix operand order repaired and checked by reading, not compiled, not run on hardware
//
#include <hip/hip_runtime.h>
#include <hip/hip_bf16.h>
#include <math.h>

// ---------------------------------------------------------------------------
// SwapGNN forward for gfx1250 (MI455X).
//   GEMMs (GAT linear + MLP) run on v_wmma_f32_16x16x32_f16 (f16 in, f32 acc).
//   Each wave computes a 16x64 output strip: A fragment reused across 4 WMMAs
//   per K-step (load:wmma ratio 2.5:1 instead of 4:1).
//   Edge softmax/aggregation uses f32 global atomics (resident in 192MB L2).
//   Sampling approximated by deterministic argmax (device-side packed atomicMax).
//
// Input index assumptions (setup_inputs dict order, params flattened with
// jax-style sorted dict keys: gats < mlp < proj_W < proj_b < type_emb;
// within each gat dict: W < a_d < a_s < b; within mlp dict: W < b):
//   0 type_ids(N) 1 requests(N) 2 edge_index(2E) 3 active_mask(N)
//   4+4i: gats[i].W, .a_d, .a_s, .b      (i=0..3)
//   20+2j: mlp[j].W, .b                  (j=0..4)
//   30 proj_W 31 proj_b 32 type_emb
// ---------------------------------------------------------------------------

typedef __attribute__((ext_vector_type(16))) _Float16 v16h;
typedef __attribute__((ext_vector_type(8)))  _Float16 v8h;
typedef __attribute__((ext_vector_type(8)))  float    v8f;

#define NN   50000
#define EE   1600000
#define ETOT (EE + NN)
#define LL   15
#define FD   15
#define HC   64
#define NH   4
#define CH   16
#define FC   256
#define NEG_SLOPE 0.2f

// ---------------- workspace layout (bytes) ----------------
#define OFF_STATS   0u          // 2 f32: sum, sumsq
#define OFF_PACK1   64u         // u64 packed argmax (removed_logits)
#define OFF_PACK2   72u         // u64 packed argmax (new_logits)
#define OFF_Q       128u        // 64 f32
#define OFF_W16     1024u
#define OFF_W0P     (OFF_W16)                   // 64x32 f16
#define OFF_W1      (OFF_W0P + 4096u)           // 64x64 f16
#define OFF_W2      (OFF_W1  + 8192u)
#define OFF_W3      (OFF_W2  + 8192u)
#define OFF_M0      (OFF_W3  + 8192u)           // 256x64 f16
#define OFF_M1      (OFF_M0  + 32768u)          // 256x256 f16
#define OFF_M2      (OFF_M1  + 131072u)
#define OFF_M3      (OFF_M2  + 131072u)
#define OFF_ACTA    (OFF_M3  + 131072u)         // N*256 f16 ping
#define OFF_ACTB    (OFF_ACTA + (size_t)NN*FC*2) // N*256 f16 pong
#define OFF_H       (OFF_ACTB + (size_t)NN*FC*2) // N*64 f32 (GAT linear out)
#define OFF_OUT     (OFF_H    + (size_t)NN*HC*4) // N*64 f32 (aggregation)
#define OFF_HL1     (OFF_OUT  + (size_t)NN*HC*4) // N*64 f32 (last GAT, kept)
#define OFF_AS      (OFF_HL1  + (size_t)NN*HC*4) // N*4 f32
#define OFF_AD      (OFF_AS   + (size_t)NN*NH*4)
#define OFF_MENC    (OFF_AD   + (size_t)NN*NH*4) // N*4 u32 (encoded max)
#define OFF_DEN     (OFF_MENC + (size_t)NN*NH*4) // N*4 f32

__device__ __forceinline__ unsigned enc_f(float f) {
  unsigned b = __float_as_uint(f);
  return (b & 0x80000000u) ? ~b : (b | 0x80000000u);
}
__device__ __forceinline__ float dec_f(unsigned u) {
  return __uint_as_float((u & 0x80000000u) ? (u & 0x7fffffffu) : ~u);
}

// ---------------- generic fills ----------------
__global__ void fill_u32(unsigned* p, unsigned v, long n) {
  long t = (long)blockIdx.x * blockDim.x + threadIdx.x;
  if (t < n) p[t] = v;
}

// ---------------- request stats ----------------
__global__ void req_stats(const float* __restrict__ req, float* stats, int cnt) {
  __shared__ float ss[256], sq[256];
  int t = blockIdx.x * blockDim.x + threadIdx.x;
  float v = (t < cnt) ? req[LL + t] : 0.f;
  ss[threadIdx.x] = v; sq[threadIdx.x] = v * v;
  __syncthreads();
  for (int s = 128; s > 0; s >>= 1) {
    if (threadIdx.x < s) { ss[threadIdx.x] += ss[threadIdx.x + s]; sq[threadIdx.x] += sq[threadIdx.x + s]; }
    __syncthreads();
  }
  if (threadIdx.x == 0) { atomicAdd(&stats[0], ss[0]); atomicAdd(&stats[1], sq[0]); }
}

// ---------------- build x0 (N x 32 f16, cols 16..31 zero) ----------------
__global__ void build_x0(const int* __restrict__ type_ids, const float* __restrict__ req,
                         const float* __restrict__ emb, const float* __restrict__ stats,
                         _Float16* __restrict__ x0) {
  int n = blockIdx.x * blockDim.x + threadIdx.x;
  if (n >= NN) return;
  float cnt  = (float)(NN - LL);
  float mean = stats[0] / cnt;
  float var  = (stats[1] - stats[0] * stats[0] / cnt) / (cnt - 1.f);
  float inv  = rsqrtf(var);
  int tid = type_ids[n];
  _Float16* row = x0 + (size_t)n * 32;
  #pragma unroll
  for (int j = 0; j < FD; j++) row[j] = (_Float16)emb[tid * FD + j];
  float rv = req[n];
  if (n >= LL) rv = (rv - mean) * inv;
  row[FD] = (_Float16)rv;
  #pragma unroll
  for (int j = 16; j < 32; j++) row[j] = (_Float16)0.f;
}

// ---------------- f32 -> f16 weight convert with K padding ----------------
__global__ void cvt_pad(const float* __restrict__ src, _Float16* __restrict__ dst,
                        int rows, int cols, int kp) {
  int t = blockIdx.x * blockDim.x + threadIdx.x;
  if (t >= rows * kp) return;
  int r = t / kp, k = t - r * kp;
  dst[t] = (k < cols) ? (_Float16)src[r * cols + k] : (_Float16)0.f;
}

// ---------------- WMMA GEMM: C[M,O] = A[M,Kp] * B[O,Kp]^T ----------------
// One wave -> one 16x64 output strip (4 WMMA tiles, A fragment reused 4x).
// 8 waves/block over consecutive M tiles; blockIdx.y covers O in steps of 64.
__global__ void gemm_wmma(const _Float16* __restrict__ A, int Kp,
                          const _Float16* __restrict__ B,
                          const float* __restrict__ bias, int relu,
                          float* __restrict__ Cf, _Float16* __restrict__ Ch,
                          int M, int O) {
  int wave  = threadIdx.x >> 5;
  int lane  = threadIdx.x & 31;
  int mTile = blockIdx.x * (blockDim.x >> 5) + wave;
  int nTiles = M >> 4;
  if (mTile >= nTiles) return;           // uniform per wave: EXEC stays all-ones
  int oBase = blockIdx.y << 6;
  int half  = lane >> 4;
  int r     = lane & 15;

  v8f c0 = {}, c1 = {}, c2 = {}, c3 = {};
  const _Float16* Arow = A + (size_t)(mTile * 16 + r) * Kp;
  const _Float16* Brow = B + (size_t)(oBase + r) * Kp;   // tile j adds j*16*Kp
  const size_t bStride = (size_t)16 * Kp;

  for (int kt = 0; kt < Kp; kt += 32) {
    // A fragment: row = mTile*16 + r; 16 contiguous halves at K = kt+half*16
    const v8h* ap = (const v8h*)(Arow + kt + half * 16);
    v8h alo = ap[0];
    v8h ahi = ap[1];
    v16h a;
    #pragma unroll
    for (int i = 0; i < 8; i++) { a[i] = alo[i]; a[8 + i] = ahi[i]; }
    // B fragments: col = oBase + j*16 + r; 16 contiguous halves at K = kt+half*16
    #pragma unroll
    for (int j = 0; j < 4; j++) {
      const v8h* bp = (const v8h*)(Brow + j * bStride + kt + half * 16);
      v8h blo = bp[0];
      v8h bhi = bp[1];
      v16h b;
      #pragma unroll
      for (int i = 0; i < 8; i++) { b[i] = blo[i]; b[8 + i] = bhi[i]; }
      if      (j == 0) c0 = __builtin_amdgcn_wmma_f32_16x16x32_f16(false, a, false, b, (short)0, c0, false, false);
      else if (j == 1) c1 = __builtin_amdgcn_wmma_f32_16x16x32_f16(false, a, false, b, (short)0, c1, false, false);
      else if (j == 2) c2 = __builtin_amdgcn_wmma_f32_16x16x32_f16(false, a, false, b, (short)0, c2, false, false);
      else             c3 = __builtin_amdgcn_wmma_f32_16x16x32_f16(false, a, false, b, (short)0, c3, false, false);
    }
  }
  #pragma unroll
  for (int j = 0; j < 4; j++) {
    v8f c = (j == 0) ? c0 : (j == 1) ? c1 : (j == 2) ? c2 : c3;
    int col = oBase + j * 16 + r;
    #pragma unroll
    for (int g = 0; g < 8; g++) {
      int row = mTile * 16 + half * 8 + g;
      float v = c[g];
      if (bias) v += bias[col];
      if (relu) v = v > 0.f ? v : 0.f;
      if (Cf) Cf[(size_t)row * O + col] = v;
      if (Ch) Ch[(size_t)row * O + col] = (_Float16)v;
    }
  }
}

// ---------------- per-node attention coefficients ----------------
__global__ void attn_coeff(const float* __restrict__ H, const float* __restrict__ pas,
                           const float* __restrict__ pad_, float* __restrict__ as_,
                           float* __restrict__ ad_) {
  int t = blockIdx.x * blockDim.x + threadIdx.x;
  if (t >= NN * NH) return;
  int n = t >> 2, hh = t & 3;
  const float* hv = H + (size_t)n * HC + hh * CH;
  float s = 0.f, d = 0.f;
  #pragma unroll
  for (int c = 0; c < CH; c++) { s += hv[c] * pas[hh * CH + c]; d += hv[c] * pad_[hh * CH + c]; }
  as_[t] = s; ad_[t] = d;
}

// ---------------- edge passes ----------------
__device__ __forceinline__ void edge_sd(const int* ei, int e, int& src, int& dst) {
  if (e < EE) { src = ei[e]; dst = ei[EE + e]; }
  else        { src = e - EE; dst = e - EE; }   // appended self loops
}
__device__ __forceinline__ float leaky(float v) { return v > 0.f ? v : NEG_SLOPE * v; }

__global__ void edge_max(const int* __restrict__ ei, const float* __restrict__ as_,
                         const float* __restrict__ ad_, unsigned* __restrict__ menc) {
  long t = (long)blockIdx.x * blockDim.x + threadIdx.x;
  if (t >= (long)ETOT * NH) return;
  int e = (int)(t >> 2), hh = (int)(t & 3), src, dst;
  edge_sd(ei, e, src, dst);
  float v = leaky(as_[src * NH + hh] + ad_[dst * NH + hh]);
  atomicMax(&menc[dst * NH + hh], enc_f(v));
}

__global__ void edge_expsum(const int* __restrict__ ei, const float* __restrict__ as_,
                            const float* __restrict__ ad_, const unsigned* __restrict__ menc,
                            float* __restrict__ den) {
  long t = (long)blockIdx.x * blockDim.x + threadIdx.x;
  if (t >= (long)ETOT * NH) return;
  int e = (int)(t >> 2), hh = (int)(t & 3), src, dst;
  edge_sd(ei, e, src, dst);
  float v = leaky(as_[src * NH + hh] + ad_[dst * NH + hh]);
  atomicAdd(&den[dst * NH + hh], __expf(v - dec_f(menc[dst * NH + hh])));
}

__global__ void edge_agg(const int* __restrict__ ei, const float* __restrict__ as_,
                         const float* __restrict__ ad_, const unsigned* __restrict__ menc,
                         const float* __restrict__ den, const float* __restrict__ H,
                         float* __restrict__ Out) {
  long t = (long)blockIdx.x * blockDim.x + threadIdx.x;
  if (t >= (long)ETOT * NH) return;
  int e = (int)(t >> 2), hh = (int)(t & 3), src, dst;
  edge_sd(ei, e, src, dst);
  float v = leaky(as_[src * NH + hh] + ad_[dst * NH + hh]);
  float alpha = __expf(v - dec_f(menc[dst * NH + hh])) / (den[dst * NH + hh] + 1e-16f);
  const float* hs = H + (size_t)src * HC + hh * CH;
  float* od = Out + (size_t)dst * HC + hh * CH;
  #pragma unroll
  for (int c = 0; c < CH; c++) atomicAdd(&od[c], alpha * hs[c]);
}

// ---------------- GAT epilogue: +bias (, relu), emit f16 (, f32) ----------------
__global__ void gat_epilogue(const float* __restrict__ Out, const float* __restrict__ bias,
                             int relu, _Float16* __restrict__ act, float* __restrict__ keep) {
  long t = (long)blockIdx.x * blockDim.x + threadIdx.x;
  if (t >= (long)NN * HC) return;
  int c = (int)(t & (HC - 1));
  float v = Out[t] + bias[c];
  if (relu) v = v > 0.f ? v : 0.f;
  act[t] = (_Float16)v;
  if (keep) keep[t] = v;
}

// ---------------- final MLP dot + remove mask + argmax ----------------
__global__ void final_dot(const _Float16* __restrict__ y, const float* __restrict__ W4,
                          const float* __restrict__ b4, const float* __restrict__ amask,
                          float* __restrict__ out_logits, unsigned long long* packed) {
  int n = blockIdx.x * blockDim.x + threadIdx.x;
  if (n >= NN) return;
  float acc = b4[0];
  const _Float16* yr = y + (size_t)n * FC;
  #pragma unroll 8
  for (int k = 0; k < FC; k++) acc += (float)yr[k] * W4[k];
  float mask;
  if (n < LL) mask = (amask[n] == 0.f) ? -INFINITY : 0.f;  // swap 0 <-> -inf
  else        mask = amask[n];
  acc += mask;
  out_logits[n] = acc;
  unsigned long long p = ((unsigned long long)enc_f(acc) << 32) | (unsigned)n;
  atomicMax(packed, p);
}

// ---------------- q = tanh(h_l1[a1] @ projW^T + projb), single block ----------------
__global__ void proj_q(const float* __restrict__ hl1, const float* __restrict__ pw,
                       const float* __restrict__ pb, const unsigned long long* packed1,
                       float* __restrict__ q) {
  int o = threadIdx.x;
  if (o >= HC) return;
  int a1 = (int)(packed1[0] & 0xffffffffull);
  const float* hr = hl1 + (size_t)a1 * HC;
  float acc = pb[o];
  #pragma unroll
  for (int k = 0; k < HC; k++) acc += hr[k] * pw[o * HC + k];
  q[o] = tanhf(acc);
}

// ---------------- new logits + argmax ----------------
__global__ void new_logits(const float* __restrict__ hl1, const float* __restrict__ q,
                           const float* __restrict__ amask, const unsigned long long* packed1,
                           float* __restrict__ out_logits, unsigned long long* packed2) {
  int n = blockIdx.x * blockDim.x + threadIdx.x;
  if (n >= NN) return;
  int a1 = (int)(packed1[0] & 0xffffffffull);
  const float* hr = hl1 + (size_t)n * HC;
  float acc = 0.f;
  #pragma unroll
  for (int k = 0; k < HC; k++) acc += hr[k] * q[k];
  acc += (n == a1) ? 0.f : amask[n];
  out_logits[n] = acc;
  unsigned long long p = ((unsigned long long)enc_f(acc) << 32) | (unsigned)n;
  atomicMax(packed2, p);
}

__global__ void write_actions(const unsigned long long* p1, const unsigned long long* p2,
                              float* out) {
  if (threadIdx.x == 0) {
    out[2 * NN + 0] = (float)(int)(p1[0] & 0xffffffffull);
    out[2 * NN + 1] = (float)(int)(p2[0] & 0xffffffffull);
  }
}

// ---------------------------------------------------------------------------
extern "C" void kernel_launch(void* const* d_in, const int* in_sizes, int n_in,
                              void* d_out, int out_size, void* d_ws, size_t ws_size,
                              hipStream_t stream) {
  (void)in_sizes; (void)n_in; (void)out_size; (void)ws_size;
  char* ws = (char*)d_ws;
  const int*   type_ids = (const int*)  d_in[0];
  const float* requests = (const float*)d_in[1];
  const int*   edge_idx = (const int*)  d_in[2];
  const float* amask    = (const float*)d_in[3];
  const float* gW[4]  = { (const float*)d_in[4],  (const float*)d_in[8],
                          (const float*)d_in[12], (const float*)d_in[16] };
  const float* gAD[4] = { (const float*)d_in[5],  (const float*)d_in[9],
                          (const float*)d_in[13], (const float*)d_in[17] };
  const float* gAS[4] = { (const float*)d_in[6],  (const float*)d_in[10],
                          (const float*)d_in[14], (const float*)d_in[18] };
  const float* gB[4]  = { (const float*)d_in[7],  (const float*)d_in[11],
                          (const float*)d_in[15], (const float*)d_in[19] };
  const float* mW[5]  = { (const float*)d_in[20], (const float*)d_in[22],
                          (const float*)d_in[24], (const float*)d_in[26],
                          (const float*)d_in[28] };
  const float* mB[5]  = { (const float*)d_in[21], (const float*)d_in[23],
                          (const float*)d_in[25], (const float*)d_in[27],
                          (const float*)d_in[29] };
  const float* projW  = (const float*)d_in[30];
  const float* projB  = (const float*)d_in[31];
  const float* temb   = (const float*)d_in[32];
  float* out = (float*)d_out;

  float*     stats = (float*)(ws + OFF_STATS);
  unsigned long long* pack1 = (unsigned long long*)(ws + OFF_PACK1);
  unsigned long long* pack2 = (unsigned long long*)(ws + OFF_PACK2);
  float*     qbuf  = (float*)(ws + OFF_Q);
  _Float16*  w16[4] = { (_Float16*)(ws + OFF_W0P), (_Float16*)(ws + OFF_W1),
                        (_Float16*)(ws + OFF_W2),  (_Float16*)(ws + OFF_W3) };
  _Float16*  m16[4] = { (_Float16*)(ws + OFF_M0),  (_Float16*)(ws + OFF_M1),
                        (_Float16*)(ws + OFF_M2),  (_Float16*)(ws + OFF_M3) };
  _Float16*  actA = (_Float16*)(ws + OFF_ACTA);
  _Float16*  actB = (_Float16*)(ws + OFF_ACTB);
  float*     Hbuf = (float*)(ws + OFF_H);
  float*     Obuf = (float*)(ws + OFF_OUT);
  float*     HL1  = (float*)(ws + OFF_HL1);
  float*     asb  = (float*)(ws + OFF_AS);
  float*     adb  = (float*)(ws + OFF_AD);
  unsigned*  menc = (unsigned*)(ws + OFF_MENC);
  float*     den  = (float*)(ws + OFF_DEN);

  const int B256 = 256;
  auto blk = [](long n) { return (unsigned)((n + 255) / 256); };

  // --- stats + x0 ---
  fill_u32<<<1, 64, 0, stream>>>((unsigned*)stats, 0u, 8);
  req_stats<<<blk(NN - LL), B256, 0, stream>>>(requests, stats, NN - LL);
  build_x0<<<blk(NN), B256, 0, stream>>>(type_ids, requests, temb, stats, actA);

  // --- weight conversion (f32 -> f16, K padded) ---
  cvt_pad<<<blk(64 * 32), B256, 0, stream>>>(gW[0], w16[0], 64, 16, 32);
  for (int i = 1; i < 4; i++)
    cvt_pad<<<blk(64 * 64), B256, 0, stream>>>(gW[i], w16[i], 64, 64, 64);
  cvt_pad<<<blk(256 * 64), B256, 0, stream>>>(mW[0], m16[0], 256, 64, 64);
  for (int j = 1; j < 4; j++)
    cvt_pad<<<blk(256 * 256), B256, 0, stream>>>(mW[j], m16[j], 256, 256, 256);

  // --- GAT layers ---
  const long ETH = (long)ETOT * NH;
  for (int layer = 0; layer < 4; layer++) {
    int Kp = (layer == 0) ? 32 : 64;
    dim3 gg((3125 + 7) / 8, HC / 64);
    gemm_wmma<<<gg, B256, 0, stream>>>(actA, Kp, w16[layer], nullptr, 0,
                                       Hbuf, nullptr, NN, HC);
    attn_coeff<<<blk((long)NN * NH), B256, 0, stream>>>(Hbuf, gAS[layer], gAD[layer], asb, adb);
    fill_u32<<<blk((long)NN * NH), B256, 0, stream>>>(menc, 0u, (long)NN * NH);
    fill_u32<<<blk((long)NN * NH), B256, 0, stream>>>((unsigned*)den, 0u, (long)NN * NH);
    fill_u32<<<blk((long)NN * HC), B256, 0, stream>>>((unsigned*)Obuf, 0u, (long)NN * HC);
    edge_max   <<<blk(ETH), B256, 0, stream>>>(edge_idx, asb, adb, menc);
    edge_expsum<<<blk(ETH), B256, 0, stream>>>(edge_idx, asb, adb, menc, den);
    edge_agg   <<<blk(ETH), B256, 0, stream>>>(edge_idx, asb, adb, menc, den, Hbuf, Obuf);
    int relu = (layer < 3) ? 1 : 0;
    gat_epilogue<<<blk((long)NN * HC), B256, 0, stream>>>(Obuf, gB[layer], relu, actA,
                                                         relu ? nullptr : HL1);
  }

  // --- MLP (WMMA) ---
  dim3 g0((3125 + 7) / 8, FC / 64);
  gemm_wmma<<<g0, B256, 0, stream>>>(actA, 64, m16[0], mB[0], 1, nullptr, actB, NN, FC);
  gemm_wmma<<<g0, B256, 0, stream>>>(actB, 256, m16[1], mB[1], 1, nullptr, actA, NN, FC);
  gemm_wmma<<<g0, B256, 0, stream>>>(actA, 256, m16[2], mB[2], 1, nullptr, actB, NN, FC);
  gemm_wmma<<<g0, B256, 0, stream>>>(actB, 256, m16[3], mB[3], 1, nullptr, actA, NN, FC);

  // --- heads / sampling (greedy argmax approximation) ---
  fill_u32<<<1, 64, 0, stream>>>((unsigned*)(void*)pack1, 0u, 4);  // pack1+pack2
  final_dot<<<blk(NN), B256, 0, stream>>>(actA, mW[4], mB[4], amask, out, pack1);
  proj_q<<<1, 64, 0, stream>>>(HL1, projW, projB, pack1, qbuf);
  new_logits<<<blk(NN), B256, 0, stream>>>(HL1, qbuf, amask, pack1, out + NN, pack2);
  write_actions<<<1, 32, 0, stream>>>(pack1, pack2, out);
}
